// GraphModel_38431367365011
// MI455X (gfx1250) — matrix operand order repaired and checked by reading, not compiled, not run on hardware
//
#include <hip/hip_runtime.h>
#include <math.h>

// ---------------------------------------------------------------------------
// GNN forward: RGCNConv (mean agg, 3 relations) -> TransformerConv (4 heads)
// -> classifier.  Dense GEMMs via v_wmma_f32_16x16x32_bf16 (wave32, gfx1250),
// sparse segment ops via float atomics.
// ---------------------------------------------------------------------------

typedef __bf16 bf16;
typedef __attribute__((ext_vector_type(16))) __bf16 v16bf;
typedef __attribute__((ext_vector_type(8)))  __bf16 v8bf;
typedef __attribute__((ext_vector_type(8)))  float  v8f;

#define GDIM  128
#define H1    64
#define HH    128   // HEADS * H2
#define HEADS 4
#define H2D   32
#define NC    7
#define NREL  3

__device__ __forceinline__ bf16 f2bf(float f) {
  unsigned u = __float_as_uint(f);
  unsigned r = u + 0x7fffu + ((u >> 16) & 1u);   // round-to-nearest-even
  unsigned short h = (unsigned short)(r >> 16);
  bf16 b;
  __builtin_memcpy(&b, &h, sizeof(h));
  return b;
}

__device__ __forceinline__ void atomicMaxF(float* addr, float v) {
  // IEEE-ordered float max via int/uint monotonic mapping
  if (v >= 0.0f) atomicMax((int*)addr, __float_as_int(v));
  else           atomicMin((unsigned int*)addr, __float_as_uint(v));
}

// ---------------------------------------------------------------------------
__global__ void k_f32_to_bf16(const float* __restrict__ in, bf16* __restrict__ out, size_t n) {
  size_t i = (size_t)blockIdx.x * blockDim.x + threadIdx.x;
  if (i < n) out[i] = f2bf(in[i]);
}

__global__ void k_fill_f32(float* __restrict__ p, float v, size_t n) {
  size_t i = (size_t)blockIdx.x * blockDim.x + threadIdx.x;
  if (i < n) p[i] = v;
}

// Pack B [K x N] row-major f32 into per-lane WMMA bf16 B-fragment order:
// Bp[(((ntile*(K/32)+kslab)*32 + lane)*16 + t] = B[k*N + col]
//   col = ntile*16 + lane%16 ; k = kslab*32 + (t/8)*16 + (lane/16)*8 + t%8
__global__ void k_pack_B(const float* __restrict__ B, bf16* __restrict__ Bp, int K, int N) {
  int idx = blockIdx.x * blockDim.x + threadIdx.x;
  if (idx >= K * N) return;
  int t      = idx & 15;
  int lane   = (idx >> 4) & 31;
  int rest   = idx >> 9;             // ntile*(K/32) + kslab
  int kslabs = K >> 5;
  int ntile  = rest / kslabs;
  int kslab  = rest - ntile * kslabs;
  int col    = ntile * 16 + (lane & 15);
  int k      = kslab * 32 + (t >> 3) * 16 + (lane >> 4) * 8 + (t & 7);
  Bp[idx] = f2bf(B[(size_t)k * N + col]);
}

// C[M x N] = A[M x K](bf16,row-major) @ Bpacked + bias, optional ReLU.
// One 16x16 output tile per wave; 8 waves per block walk M.
__global__ void __launch_bounds__(256)
k_wmma_gemm(const bf16* __restrict__ A, const bf16* __restrict__ Bp,
            const float* __restrict__ bias, float* __restrict__ C,
            int Mtiles, int K, int N, int relu) {
  int wave  = threadIdx.x >> 5;
  int lane  = threadIdx.x & 31;
  int mtile = blockIdx.x * 8 + wave;
  if (mtile >= Mtiles) return;                 // wave-uniform: EXEC stays all-1
  int ntile = blockIdx.y;
  int hi    = lane >> 4;                       // 0 or 1 (lane group)
  int l15   = lane & 15;
  int row   = mtile * 16 + l15;
  int kslabs = K >> 5;

  const bf16* arow = A + (size_t)row * K + hi * 8;
  const bf16* bp   = Bp + (((size_t)ntile * kslabs) * 32 + lane) * 16;

  v8f acc = {};
  for (int ks = 0; ks < kslabs; ++ks) {
    // A 16x32 bf16 fragment: elems 0..7 = K base..base+7, 8..15 = K base+16..base+23
    v8bf alo = *(const v8bf*)(arow);
    v8bf ahi = *(const v8bf*)(arow + 16);
    v16bf a = __builtin_shufflevector(alo, ahi, 0,1,2,3,4,5,6,7,8,9,10,11,12,13,14,15);
    v16bf b = *(const v16bf*)bp;
    acc = __builtin_amdgcn_wmma_f32_16x16x32_bf16(
        /*neg_a=*/false, a, /*neg_b=*/false, b,
        /*c_mod=*/(short)0, acc, /*reuse_a=*/false, /*reuse_b=*/false);
    arow += 32;
    bp   += 32 * 16;
  }

  int col = ntile * 16 + l15;
  float bv = bias ? bias[col] : 0.0f;
  // C layout: elem i -> row-in-tile = i + 8*hi, col = lane&15
  float* crow = C + (size_t)(mtile * 16 + hi * 8) * N + col;
#pragma unroll
  for (int i = 0; i < 8; ++i) {
    float v = acc[i] + bv;
    if (relu) v = fmaxf(v, 0.0f);
    crow[(size_t)i * N] = v;
  }
}

// ---------------- RGCN sparse part ----------------
__global__ void k_rgcn_scatter(const int* __restrict__ src, const int* __restrict__ dst,
                               const int* __restrict__ etype, const float* __restrict__ HR,
                               float* __restrict__ ssum, float* __restrict__ cnt,
                               int E, int n) {
  size_t idx = (size_t)blockIdx.x * blockDim.x + threadIdx.x;
  if (idx >= (size_t)E * H1) return;
  int e = (int)(idx >> 6);
  int c = (int)(idx & 63);
  int r = etype[e];
  int s = src[e], d = dst[e];
  atomicAdd(&ssum[((size_t)r * n + d) * H1 + c], HR[((size_t)r * n + s) * H1 + c]);
  if (c == 0) atomicAdd(&cnt[(size_t)r * n + d], 1.0f);
}

__global__ void k_rgcn_combine(const float* __restrict__ h0, const float* __restrict__ ssum,
                               const float* __restrict__ cnt, bf16* __restrict__ hbf, int n) {
  size_t idx = (size_t)blockIdx.x * blockDim.x + threadIdx.x;
  if (idx >= (size_t)n * H1) return;
  int node = (int)(idx >> 6);
  int c    = (int)(idx & 63);
  float v = h0[idx];
#pragma unroll
  for (int r = 0; r < NREL; ++r) {
    float ct = cnt[(size_t)r * n + node];
    if (ct > 0.0f) v += ssum[((size_t)r * n + node) * H1 + c] / ct;
  }
  v = fmaxf(v, 0.0f);
  hbf[idx] = f2bf(v);
}

// ---------------- TransformerConv sparse part ----------------
__global__ void k_attn_scores(const int* __restrict__ src, const int* __restrict__ dst,
                              const float* __restrict__ Q, const float* __restrict__ Km,
                              float* __restrict__ scores, float* __restrict__ smax, int E) {
  size_t idx = (size_t)blockIdx.x * blockDim.x + threadIdx.x;
  if (idx >= (size_t)E * HEADS) return;
  int e = (int)(idx >> 2);
  int h = (int)(idx & 3);
  int d = dst[e], s = src[e];
  const float4* q = (const float4*)(Q  + (size_t)d * HH + h * H2D);
  const float4* k = (const float4*)(Km + (size_t)s * HH + h * H2D);
  float acc = 0.0f;
#pragma unroll
  for (int i = 0; i < 8; ++i) {
    float4 qa = q[i], kb = k[i];
    acc += qa.x * kb.x + qa.y * kb.y + qa.z * kb.z + qa.w * kb.w;
  }
  acc *= 0.17677669529663687f;   // 1/sqrt(32)
  scores[idx] = acc;
  atomicMaxF(&smax[(size_t)d * HEADS + h], acc);
}

__global__ void k_attn_exp(const int* __restrict__ dst, float* __restrict__ scores,
                           const float* __restrict__ smax, float* __restrict__ denom, int E) {
  size_t idx = (size_t)blockIdx.x * blockDim.x + threadIdx.x;
  if (idx >= (size_t)E * HEADS) return;
  int e = (int)(idx >> 2);
  int h = (int)(idx & 3);
  int d = dst[e];
  float ev = __expf(scores[idx] - smax[(size_t)d * HEADS + h]);
  scores[idx] = ev;                 // becomes softmax numerator
  atomicAdd(&denom[(size_t)d * HEADS + h], ev);
}

__global__ void k_attn_agg(const int* __restrict__ src, const int* __restrict__ dst,
                           const float* __restrict__ num, const float* __restrict__ denom,
                           const float* __restrict__ V, float* __restrict__ agg, int E) {
  size_t idx = (size_t)blockIdx.x * blockDim.x + threadIdx.x;
  if (idx >= (size_t)E * HH) return;
  int e = (int)(idx >> 7);
  int c = (int)(idx & 127);
  int h = c >> 5;
  int d = dst[e], s = src[e];
  float alpha = num[(size_t)e * HEADS + h] / fmaxf(denom[(size_t)d * HEADS + h], 1e-16f);
  atomicAdd(&agg[(size_t)d * HH + c], alpha * V[(size_t)s * HH + c]);
}

// out = relu(agg + skip) @ Wf + bf
__global__ void k_final(const float* __restrict__ agg, const float* __restrict__ SK,
                        const float* __restrict__ Wf, const float* __restrict__ bfv,
                        float* __restrict__ out, int n) {
  int node = blockIdx.x * blockDim.x + threadIdx.x;
  if (node >= n) return;
  float acc[NC];
#pragma unroll
  for (int j = 0; j < NC; ++j) acc[j] = bfv[j];
  const float* a  = agg + (size_t)node * HH;
  const float* sk = SK  + (size_t)node * HH;
  for (int k = 0; k < HH; ++k) {
    float ov = fmaxf(a[k] + sk[k], 0.0f);
#pragma unroll
    for (int j = 0; j < NC; ++j) acc[j] += ov * Wf[k * NC + j];
  }
#pragma unroll
  for (int j = 0; j < NC; ++j) out[(size_t)node * NC + j] = acc[j];
}

// ---------------------------------------------------------------------------
extern "C" void kernel_launch(void* const* d_in, const int* in_sizes, int n_in,
                              void* d_out, int out_size, void* d_ws, size_t ws_size,
                              hipStream_t stream) {
  (void)n_in; (void)out_size; (void)ws_size;

  const float* x      = (const float*)d_in[0];
  const int* edge_idx = (const int*)d_in[2];
  const int* etype    = (const int*)d_in[3];
  const float* W_rel  = (const float*)d_in[4];
  const float* W_root = (const float*)d_in[5];
  const float* b_rgcn = (const float*)d_in[6];
  const float* Wq = (const float*)d_in[7],  *bq  = (const float*)d_in[8];
  const float* Wk = (const float*)d_in[9],  *bk  = (const float*)d_in[10];
  const float* Wv = (const float*)d_in[11], *bv  = (const float*)d_in[12];
  const float* Ws = (const float*)d_in[13], *bs  = (const float*)d_in[14];
  const float* Wf = (const float*)d_in[15], *bf_ = (const float*)d_in[16];
  float* out = (float*)d_out;

  const int n = in_sizes[0] / GDIM;     // 100000 (divisible by 16)
  const int E = in_sizes[2] / 2;        // 1600000
  const int Mtiles = n / 16;
  const int* src = edge_idx;            // edge_index[0]
  const int* dst = edge_idx + E;        // edge_index[1]

  // ---- workspace bump allocator (256B-aligned carves) ----
  char* p = (char*)d_ws;
  auto carve = [&](size_t bytes) -> void* {
    void* r = (void*)p;
    p += (bytes + 255) & ~(size_t)255;
    return r;
  };
  bf16* Xbf = (bf16*)carve((size_t)n * GDIM * sizeof(bf16));
  bf16* Bp_root = (bf16*)carve((size_t)GDIM * H1 * sizeof(bf16));
  bf16* Bp_rel[NREL];
  for (int r = 0; r < NREL; ++r) Bp_rel[r] = (bf16*)carve((size_t)GDIM * H1 * sizeof(bf16));
  bf16* Bp_q  = (bf16*)carve((size_t)H1 * HH * sizeof(bf16));
  bf16* Bp_k  = (bf16*)carve((size_t)H1 * HH * sizeof(bf16));
  bf16* Bp_v  = (bf16*)carve((size_t)H1 * HH * sizeof(bf16));
  bf16* Bp_s  = (bf16*)carve((size_t)H1 * HH * sizeof(bf16));
  bf16* hbf   = (bf16*)carve((size_t)n * H1 * sizeof(bf16));
  float* SKp   = (float*)carve((size_t)n * HH * sizeof(float));
  float* scores= (float*)carve((size_t)E * HEADS * sizeof(float));
  float* smax  = (float*)carve((size_t)n * HEADS * sizeof(float));
  float* denom = (float*)carve((size_t)n * HEADS * sizeof(float));
  float* agg   = (float*)carve((size_t)n * HH * sizeof(float));
  // Union region: RGCN scratch {h0,HR,ssum,cnt} reused later as {Q,K,V}
  char* u = p;
  float* h0   = (float*)u;
  float* HR   = h0 + (size_t)n * H1;              // 3*n*H1
  float* ssum = HR + (size_t)NREL * n * H1;       // 3*n*H1
  float* cnt  = ssum + (size_t)NREL * n * H1;     // 3*n
  float* Q    = (float*)u;
  float* Km   = Q  + (size_t)n * HH;
  float* V    = Km + (size_t)n * HH;

  auto cdiv = [](size_t a, size_t b) -> unsigned { return (unsigned)((a + b - 1) / b); };

  // 1) conversions + weight packing
  k_f32_to_bf16<<<cdiv((size_t)n * GDIM, 256), 256, 0, stream>>>(x, Xbf, (size_t)n * GDIM);
  k_pack_B<<<cdiv(GDIM * H1, 256), 256, 0, stream>>>(W_root, Bp_root, GDIM, H1);
  for (int r = 0; r < NREL; ++r)
    k_pack_B<<<cdiv(GDIM * H1, 256), 256, 0, stream>>>(W_rel + (size_t)r * GDIM * H1, Bp_rel[r], GDIM, H1);
  k_pack_B<<<cdiv(H1 * HH, 256), 256, 0, stream>>>(Wq, Bp_q, H1, HH);
  k_pack_B<<<cdiv(H1 * HH, 256), 256, 0, stream>>>(Wk, Bp_k, H1, HH);
  k_pack_B<<<cdiv(H1 * HH, 256), 256, 0, stream>>>(Wv, Bp_v, H1, HH);
  k_pack_B<<<cdiv(H1 * HH, 256), 256, 0, stream>>>(Ws, Bp_s, H1, HH);

  // 2) zero RGCN accumulators (fresh every call: graph replay)
  k_fill_f32<<<cdiv((size_t)NREL * n * H1, 256), 256, 0, stream>>>(ssum, 0.0f, (size_t)NREL * n * H1);
  k_fill_f32<<<cdiv((size_t)NREL * n, 256), 256, 0, stream>>>(cnt, 0.0f, (size_t)NREL * n);

  // 3) RGCN dense: h0 = x@W_root + b ; HR[r] = x@W_rel[r]
  dim3 g1(cdiv(Mtiles, 8), H1 / 16);
  k_wmma_gemm<<<g1, 256, 0, stream>>>(Xbf, Bp_root, b_rgcn, h0, Mtiles, GDIM, H1, 0);
  for (int r = 0; r < NREL; ++r)
    k_wmma_gemm<<<g1, 256, 0, stream>>>(Xbf, Bp_rel[r], nullptr, HR + (size_t)r * n * H1,
                                        Mtiles, GDIM, H1, 0);

  // 4) RGCN sparse: per-relation mean aggregation, then relu -> hbf
  k_rgcn_scatter<<<cdiv((size_t)E * H1, 256), 256, 0, stream>>>(src, dst, etype, HR, ssum, cnt, E, n);
  k_rgcn_combine<<<cdiv((size_t)n * H1, 256), 256, 0, stream>>>(h0, ssum, cnt, hbf, n);

  // 5) TransformerConv dense: Q,K,V,skip (overwrites dead RGCN scratch)
  dim3 g2(cdiv(Mtiles, 8), HH / 16);
  k_wmma_gemm<<<g2, 256, 0, stream>>>(hbf, Bp_q, bq, Q,   Mtiles, H1, HH, 0);
  k_wmma_gemm<<<g2, 256, 0, stream>>>(hbf, Bp_k, bk, Km,  Mtiles, H1, HH, 0);
  k_wmma_gemm<<<g2, 256, 0, stream>>>(hbf, Bp_v, bv, V,   Mtiles, H1, HH, 0);
  k_wmma_gemm<<<g2, 256, 0, stream>>>(hbf, Bp_s, bs, SKp, Mtiles, H1, HH, 0);

  // 6) attention softmax + aggregation
  k_fill_f32<<<cdiv((size_t)n * HEADS, 256), 256, 0, stream>>>(smax, -INFINITY, (size_t)n * HEADS);
  k_fill_f32<<<cdiv((size_t)n * HEADS, 256), 256, 0, stream>>>(denom, 0.0f, (size_t)n * HEADS);
  k_fill_f32<<<cdiv((size_t)n * HH, 256), 256, 0, stream>>>(agg, 0.0f, (size_t)n * HH);
  k_attn_scores<<<cdiv((size_t)E * HEADS, 256), 256, 0, stream>>>(src, dst, Q, Km, scores, smax, E);
  k_attn_exp<<<cdiv((size_t)E * HEADS, 256), 256, 0, stream>>>(dst, scores, smax, denom, E);
  k_attn_agg<<<cdiv((size_t)E * HH, 256), 256, 0, stream>>>(src, dst, scores, denom, V, agg, E);

  // 7) final: relu(agg + skip) @ Wf + bf
  k_final<<<cdiv(n, 128), 128, 0, stream>>>(agg, SKp, Wf, bf_, out, n);
}
